// NormFeedForwardNetworkGauntSelfTensorProductAdd_42305427866211
// MI455X (gfx1250) — matrix operand order repaired and checked by reading, not compiled, not run on hardware
//
#include <hip/hip_runtime.h>

#define NFEAT 3200      // 25*128
#define CCH 128
#define L2C 25
#define LPAD 32
#define TPN 512
#define CHUNK 128
#define PI_D 3.14159265358979323846

typedef __attribute__((ext_vector_type(16))) __bf16 v16bf;
typedef __attribute__((ext_vector_type(8)))  __bf16 v8bf;
typedef __attribute__((ext_vector_type(8)))  float  v8f;

__device__ __forceinline__ int lof(int i) {
    return (i >= 16) ? 4 : (i >= 9) ? 3 : (i >= 4) ? 2 : (i >= 1) ? 1 : 0;
}

__device__ __forceinline__ v8f zero8() {
    v8f d;
#pragma unroll
    for (int r = 0; r < 8; ++r) d[r] = 0.0f;
    return d;
}

// CDNA5 bf16 A-operand (16x32, M=rows, K=cols), src row-major, ld in elements.
// lane<16 : elems 0..7 -> K=0..7,  elems 8..15 -> K=16..23  (row M = lane)
// lane>=16: elems 0..7 -> K=8..15, elems 8..15 -> K=24..31  (row M = lane-16)
__device__ __forceinline__ v16bf load_a_tile(const __bf16* base, int ld) {
    int lane = threadIdx.x & 31;
    int half = lane >> 4;
    int row  = lane & 15;
    const __bf16* p = base + row * ld;
    v16bf a;
#pragma unroll
    for (int j = 0; j < 8; ++j) a[j] = p[half * 8 + j];
#pragma unroll
    for (int j = 0; j < 8; ++j) a[8 + j] = p[16 + half * 8 + j];
    return a;
}

// CDNA5 bf16 B-operand (32x16, K=rows, N=cols) read from TRANSPOSED [n][k] src.
// lane<16: elems 0..7 -> K=0..7, elems 8..15 -> K=16..23; lane>=16: K=8..15, K=24..31 (N=lane%16).
__device__ __forceinline__ v16bf load_b_tile(const __bf16* baseT, int ld) {
    int lane = threadIdx.x & 31;
    int half = lane >> 4;
    int col  = lane & 15;
    const __bf16* p = baseT + col * ld;
    v16bf b;
#pragma unroll
    for (int j = 0; j < 8; ++j) b[j] = p[half * 8 + j];
#pragma unroll
    for (int j = 0; j < 8; ++j) b[8 + j] = p[16 + half * 8 + j];
    return b;
}

__device__ __forceinline__ v8f wmma_bf16(v16bf a, v16bf b, v8f c) {
    return __builtin_amdgcn_wmma_f32_16x16x32_bf16(
        false, a, false, b, (short)0, c, false, false);
}

// pack 8 f32 accumulator lanes -> 8 bf16, single 16-byte store
__device__ __forceinline__ void store8_bf16(__bf16* dst, v8f v) {
    *reinterpret_cast<v8bf*>(dst) = __builtin_convertvector(v, v8bf);
}

// 8 consecutive f32, single 32-byte store
__device__ __forceinline__ void store8_f32(float* dst, v8f v) {
    *reinterpret_cast<v8f*>(dst) = v;
}

// ---------------- setup: spherical harmonics tables (match reference) --------
__global__ void build_sh_kernel(__bf16* __restrict__ Ybf, __bf16* __restrict__ YwTbf) {
    int tp = blockIdx.x * blockDim.x + threadIdx.x;
    if (tp >= TPN) return;
    int t = tp >> 5;   // theta index, ascending Gauss-Legendre nodes
    int p = tp & 31;

    // Newton for GL node (k = 16 - t gives ascending order)
    double kk = (double)(16 - t);
    double xg = cos(PI_D * (kk - 0.25) / 16.5);
    double dp = 0.0;
#pragma unroll 1
    for (int it = 0; it < 10; ++it) {
        double p0 = 1.0, p1 = xg;
        for (int j = 2; j <= 16; ++j) {
            double p2 = ((2.0 * j - 1.0) * xg * p1 - (j - 1.0) * p0) / (double)j;
            p0 = p1; p1 = p2;
        }
        dp = 16.0 * (xg * p1 - p0) / (xg * xg - 1.0);
        xg = xg - p1 / dp;
    }
    double ct = xg;
    double wt = 2.0 / ((1.0 - xg * xg) * dp * dp);
    double st = sqrt(fmax(0.0, 1.0 - ct * ct));

    double P[5][5];
    P[0][0] = 1.0;
    for (int m = 1; m <= 4; ++m) P[m][m] = -(2.0 * m - 1.0) * st * P[m - 1][m - 1];
    for (int m = 0; m < 4; ++m)  P[m + 1][m] = (2.0 * m + 1.0) * ct * P[m][m];
    for (int m = 0; m <= 4; ++m)
        for (int l = m + 2; l <= 4; ++l)
            P[l][m] = ((2.0 * l - 1.0) * ct * P[l - 1][m] - (double)(l + m - 1) * P[l - 2][m])
                      / (double)(l - m);

    double fact[9];
    fact[0] = 1.0;
    for (int j = 1; j <= 8; ++j) fact[j] = fact[j - 1] * j;

    double phi = (double)p * (2.0 * PI_D / 32.0);
    double qw  = wt * (2.0 * PI_D / 32.0);

    for (int l = 0; l <= 4; ++l) {
        for (int m = -l; m <= l; ++m) {
            int am = m < 0 ? -m : m;
            double Nn = sqrt((2.0 * l + 1.0) / (4.0 * PI_D) * fact[l - am] / fact[l + am]);
            double ang, pref;
            if (m == 0)      { ang = 1.0;           pref = 1.0; }
            else if (m > 0)  { ang = cos(m * phi);  pref = sqrt(2.0); }
            else             { ang = sin(am * phi); pref = sqrt(2.0); }
            double Yv = pref * Nn * P[l][am] * ang;
            int idx = l * l + l + m;
            Ybf[tp * LPAD + idx]  = (__bf16)(float)Yv;
            YwTbf[idx * TPN + tp] = (__bf16)(float)(Yv * qw);
        }
    }
    for (int idx = L2C; idx < LPAD; ++idx) {
        Ybf[tp * LPAD + idx]  = (__bf16)0.0f;
        YwTbf[idx * TPN + tp] = (__bf16)0.0f;
    }
}

// transpose+convert per-degree weights to bf16 [l][c][d]
__global__ void convert_weights_kernel(const float* __restrict__ w1,
                                       const float* __restrict__ w2,
                                       __bf16* __restrict__ w1T,
                                       __bf16* __restrict__ w2T) {
    int idx = blockIdx.x * blockDim.x + threadIdx.x;
    if (idx >= 5 * CCH * CCH) return;
    int l = idx >> 14;
    int c = (idx >> 7) & 127;
    int d = idx & 127;
    w1T[idx] = (__bf16)w1[l * 16384 + d * CCH + c];
    w2T[idx] = (__bf16)w2[l * 16384 + d * CCH + c];
}

// ---------------- fused per-node kernel -------------------------------------
// LDS map (byte offsets), phase-aliased:
//   [     0,  8192) sh_h    [32][128] bf16   (norm out, A op)
//   [  8192, 16384) sh_h1T  [128][32] bf16   (h1^T, B op for grid fwd)
//   [ 16384, 81920) sh_Y    [512][32] bf16 + sh_YwT [32][512] bf16  (phases B..C)
//                   sh_ofT  [128][80] f32                            (phase E)
//   [ 81920,114688) sh_f2T  [128][128] bf16  (phase C)
//                   sh_h1S  [128][80] bf16   (phase B staging)
//   [114688,122880) sh_y    [32][128] bf16
//   [122880,135680) sh_xf   [25][128] f32
//   [135680,136704) sh_red  [256] f32
__launch_bounds__(256)
__global__ void gaunt_fused_kernel(const float* __restrict__ x,
                                   const float* __restrict__ norm_w,
                                   const float* __restrict__ norm_b,
                                   const float* __restrict__ b1,
                                   const float* __restrict__ tp_w,
                                   const float* __restrict__ b2,
                                   const __bf16* __restrict__ Ybf,
                                   const __bf16* __restrict__ YwTbf,
                                   const __bf16* __restrict__ w1T,
                                   const __bf16* __restrict__ w2T,
                                   float* __restrict__ out) {
    extern __shared__ char smem[];
    __bf16* sh_h   = (__bf16*)(smem);
    __bf16* sh_h1T = (__bf16*)(smem + 8192);
    __bf16* sh_Y   = (__bf16*)(smem + 16384);
    __bf16* sh_YwT = (__bf16*)(smem + 49152);
    float*  sh_ofT = (float*)(smem + 16384);     // aliases Y/YwT (used after phase C)
    __bf16* sh_f2T = (__bf16*)(smem + 81920);
    __bf16* sh_h1S = (__bf16*)(smem + 81920);    // aliases f2T (used before phase C)
    __bf16* sh_y   = (__bf16*)(smem + 114688);
    float*  sh_xf  = (float*)(smem + 122880);
    float*  sh_red = (float*)(smem + 135680);

    int n    = blockIdx.x;
    int tid  = threadIdx.x;
    int wave = tid >> 5;
    int lane = tid & 31;
    int half = lane >> 4;
    int col  = lane & 15;
    const float* xn = x + (size_t)n * NFEAT;

    // ---- stage constants into LDS, zero sh_h padding rows ----
    {
        const unsigned int* ys = (const unsigned int*)Ybf;
        unsigned int* yd = (unsigned int*)sh_Y;
        for (int i = tid; i < TPN * LPAD / 2; i += 256) yd[i] = ys[i];
        const unsigned int* ws2 = (const unsigned int*)YwTbf;
        unsigned int* wd = (unsigned int*)sh_YwT;
        for (int i = tid; i < LPAD * TPN / 2; i += 256) wd[i] = ws2[i];
        unsigned int* hp = (unsigned int*)(sh_h + L2C * CCH);
        for (int i = tid; i < (LPAD - L2C) * CCH / 2; i += 256) hp[i] = 0u;
    }

    // ---- Phase A: EquivariantMergeLayerNorm ----
    for (int i = tid; i < NFEAT; i += 256) sh_xf[i] = xn[i];
    __syncthreads();

    sh_red[tid] = (tid < CCH) ? sh_xf[tid] : 0.0f;
    __syncthreads();
    for (int off = 128; off > 0; off >>= 1) {
        if (tid < off) sh_red[tid] += sh_red[tid + off];
        __syncthreads();
    }
    float mean0 = sh_red[0] * (1.0f / 128.0f);
    __syncthreads();
    if (tid < CCH) sh_xf[tid] -= mean0;
    __syncthreads();

    float vs = 0.0f;
    for (int i = tid; i < NFEAT; i += 256) { float v = sh_xf[i]; vs += v * v; }
    sh_red[tid] = vs;
    __syncthreads();
    for (int off = 128; off > 0; off >>= 1) {
        if (tid < off) sh_red[tid] += sh_red[tid + off];
        __syncthreads();
    }
    float rs = rsqrtf(sh_red[0] * (1.0f / (float)NFEAT) + 1e-5f);

    for (int i = tid; i < NFEAT; i += 256) {
        int row = i >> 7, c = i & 127;
        float hv = sh_xf[i] * rs * norm_w[lof(row) * CCH + c];
        if (row == 0) hv += norm_b[c];
        sh_h[i] = (__bf16)hv;
    }
    __syncthreads();

    // ---- Phase B: SO3Linear 1 -> staging (disjoint 16-row slots per degree) ----
    for (int job = wave; job < 40; job += 8) {
        int l = job >> 3, nt = job & 7;
        int m0 = l * l, n0 = nt * 16;
        float b1v = b1[n0 + col];
        v8f d = zero8();
#pragma unroll
        for (int kk = 0; kk < 4; ++kk) {
            v16bf a = load_a_tile(sh_h + m0 * CCH + kk * 32, CCH);
            v16bf b = load_b_tile(w1T + l * 16384 + n0 * CCH + kk * 32, CCH);
            d = wmma_bf16(a, b, d);
        }
        d[0] += ((l == 0) & (half == 0)) ? b1v : 0.0f;   // bias on (l=0, row 0)
        store8_bf16(&sh_h1S[(n0 + col) * 80 + l * 16 + 8 * half], d);
    }
    __syncthreads();
    // repack staging -> sh_h1T [c][32] (writes zero padding for i=25..31)
    for (int idx = tid; idx < CCH * LPAD; idx += 256) {
        int c = idx >> 5, i = idx & 31;
        __bf16 v = (__bf16)0.0f;
        if (i < L2C) {
            int l = lof(i);
            v = sh_h1S[c * 80 + l * 16 + (i - l * l)];
        }
        sh_h1T[idx] = v;
    }
    __syncthreads();

    // ---- Phase C: f = Y @ h1 ; square ; y += Yw^T @ f^2 (chunked over grid) ----
    v8f accy[2];
    accy[0] = zero8();
    accy[1] = zero8();
    for (int cc = 0; cc < 4; ++cc) {
        for (int job = wave; job < 64; job += 8) {
            int mt = job >> 3, nt = job & 7;
            int m0 = cc * CHUNK + mt * 16, n0 = nt * 16;
            v16bf a = load_a_tile(sh_Y + m0 * LPAD, LPAD);
            v16bf b = load_b_tile(sh_h1T + n0 * LPAD, LPAD);
            v8f f = wmma_bf16(a, b, zero8());
            v8f f2;
#pragma unroll
            for (int r = 0; r < 8; ++r) f2[r] = f[r] * f[r];
            store8_bf16(&sh_f2T[(n0 + col) * CHUNK + mt * 16 + 8 * half], f2);
        }
        __syncthreads();
#pragma unroll
        for (int jj = 0; jj < 2; ++jj) {
            int job = wave * 2 + jj;
            int mt = job >> 3, nt = job & 7;
#pragma unroll
            for (int kk = 0; kk < 4; ++kk) {
                v16bf a = load_a_tile(sh_YwT + (mt * 16) * TPN + cc * CHUNK + kk * 32, TPN);
                v16bf b = load_b_tile(sh_f2T + (nt * 16) * CHUNK + kk * 32, CHUNK);
                accy[jj] = wmma_bf16(a, b, accy[jj]);
            }
        }
        __syncthreads();
    }

    // ---- Phase D: per-(l,c) path weight -> sh_y [i][c] ----
#pragma unroll
    for (int jj = 0; jj < 2; ++jj) {
        int job = wave * 2 + jj;
        int mt = job >> 3, nt = job & 7;
#pragma unroll
        for (int r = 0; r < 8; ++r) {
            int i = mt * 16 + r + 8 * half;
            int c = nt * 16 + col;
            int l = lof(i < L2C ? i : (L2C - 1));
            sh_y[i * CCH + c] = (__bf16)(accy[jj][r] * tp_w[l * CCH + c]);
        }
    }
    __syncthreads();

    // ---- Phase E: SO3Linear 2 -> staging sh_ofT [c][80] f32 ----
    for (int job = wave; job < 40; job += 8) {
        int l = job >> 3, nt = job & 7;
        int m0 = l * l, n0 = nt * 16;
        v8f d = zero8();
#pragma unroll
        for (int kk = 0; kk < 4; ++kk) {
            v16bf a = load_a_tile(sh_y + m0 * CCH + kk * 32, CCH);
            v16bf b = load_b_tile(w2T + l * 16384 + n0 * CCH + kk * 32, CCH);
            d = wmma_bf16(a, b, d);
        }
        store8_f32(&sh_ofT[(n0 + col) * 80 + l * 16 + 8 * half], d);
    }
    __syncthreads();

    // ---- write-out: compact rows + b2 + residual, coalesced global b32 ----
    for (int e = tid; e < NFEAT; e += 256) {
        int i = e >> 7, c = e & 127;
        int l = lof(i);
        float v = sh_ofT[c * 80 + l * 16 + (i - l * l)];
        v += (i == 0) ? b2[c] : 0.0f;
        out[(size_t)n * NFEAT + e] = v + xn[e];
    }
}

extern "C" void kernel_launch(void* const* d_in, const int* in_sizes, int n_in,
                              void* d_out, int out_size, void* d_ws, size_t ws_size,
                              hipStream_t stream) {
    (void)n_in; (void)out_size; (void)ws_size;
    const float* x      = (const float*)d_in[0];
    // d_in[1] = batch (unused; drop_path rate 0)
    const float* norm_w = (const float*)d_in[2];
    const float* norm_b = (const float*)d_in[3];
    const float* w1     = (const float*)d_in[4];
    const float* b1     = (const float*)d_in[5];
    const float* tp_w   = (const float*)d_in[6];
    const float* w2     = (const float*)d_in[7];
    const float* b2     = (const float*)d_in[8];
    float* out = (float*)d_out;

    char* ws = (char*)d_ws;
    __bf16* Ybf   = (__bf16*)(ws);                     // 512*32*2  = 32768 B
    __bf16* YwTbf = (__bf16*)(ws + 32768);             // 32*512*2  = 32768 B
    __bf16* w1T   = (__bf16*)(ws + 65536);             // 81920*2   = 163840 B
    __bf16* w2T   = (__bf16*)(ws + 65536 + 163840);    // 81920*2   = 163840 B

    int N = in_sizes[0] / NFEAT;

    build_sh_kernel<<<2, 256, 0, stream>>>(Ybf, YwTbf);
    convert_weights_kernel<<<(5 * CCH * CCH + 255) / 256, 256, 0, stream>>>(w1, w2, w1T, w2T);

    size_t smem = 136704;   // see LDS map above; < 160 KB -> 2 workgroups per WGP
    gaunt_fused_kernel<<<N, 256, smem, stream>>>(x, norm_w, norm_b, b1, tp_w, b2,
                                                 Ybf, YwTbf, w1T, w2T, out);
}
